// EdgeFtLayer_30605936951711
// MI455X (gfx1250) — matrix operand order repaired and checked by reading, not compiled, not run on hardware
//
#include <hip/hip_runtime.h>
#include <stdint.h>

// ---------------------------------------------------------------------------
// GAT-style edge layer for MI455X (gfx1250, wave32, WMMA).
// bf16 WMMA 16x16x32 with f32 accumulation for all GEMMs; segment softmax via
// monotonic-uint atomicMax + f32 atomicAdd into L2-resident node accumulators.
// ---------------------------------------------------------------------------

typedef __attribute__((ext_vector_type(16))) __bf16 v16bf;
typedef __attribute__((ext_vector_type(8)))  float  v8f;
typedef __attribute__((ext_vector_type(4)))  __bf16 v4bf;

union ABu { v16bf v; uint4 q[2]; };

#define NNODES 50000
#define NEDGES 800000
#define NODE_ELEMS (NNODES * 64)   // 3,200,000

// order-preserving float -> uint encoding for atomicMax-based segment max
__device__ __forceinline__ unsigned enc_f32(float f) {
  unsigned u = __float_as_uint(f);
  return (u & 0x80000000u) ? ~u : (u | 0x80000000u);
}
__device__ __forceinline__ float dec_f32(unsigned u) {
  return __uint_as_float((u & 0x80000000u) ? (u & 0x7fffffffu) : ~u);
}

// ---------------------------------------------------------------------------
// prep kernels
// ---------------------------------------------------------------------------
__global__ __launch_bounds__(256) void k_init(unsigned* __restrict__ mx,
                                              float* __restrict__ dn,
                                              float* __restrict__ nm) {
  const int i = blockIdx.x * blockDim.x + threadIdx.x;
  if (i < NODE_ELEMS) { mx[i] = 0u; dn[i] = 0.0f; nm[i] = 0.0f; }
}

__global__ __launch_bounds__(256) void k_cvt(const float* __restrict__ in,
                                             __bf16* __restrict__ out, long n) {
  const long i = ((long)blockIdx.x * blockDim.x + threadIdx.x) * 4;
  if (i < n) {
    const float4 f = *(const float4*)(in + i);
    v4bf o; o.x = (__bf16)f.x; o.y = (__bf16)f.y; o.z = (__bf16)f.z; o.w = (__bf16)f.w;
    *(v4bf*)(out + i) = o;
  }
}

// Pack weights (K x 64, row-major f32) into bf16 WMMA B-fragment order:
//   P[((kt*2+hi)*64 + n)*16 + kk] = W[kt*32 + hi*16 + kk][n]
// so each lane's 16 contiguous bf16 = its column's 16 consecutive K values.
__global__ __launch_bounds__(256) void k_pack(const float* __restrict__ Wa,
                                              const float* __restrict__ Wt,
                                              const float* __restrict__ We,
                                              const float* __restrict__ Wee,
                                              __bf16* __restrict__ WaP,
                                              __bf16* __restrict__ WtP,
                                              __bf16* __restrict__ WeP,
                                              __bf16* __restrict__ WeeP) {
  const int i = blockIdx.x * blockDim.x + threadIdx.x;
  if (i >= 20480) return;
  const int kk = i & 15, n = (i >> 4) & 63, g = i >> 10;
  const int k = (g >> 1) * 32 + (g & 1) * 16 + kk;
  WaP[i] = (__bf16)Wa[k * 64 + n];            // 320x64
  WtP[i] = (__bf16)Wt[k * 64 + n];            // 320x64
  if (i < 8192) WeP[i]  = (__bf16)We[k * 64 + n];   // 128x64
  if (i < 4096) WeeP[i] = (__bf16)Wee[k * 64 + n];  //  64x64
}

// ---------------------------------------------------------------------------
// Edge pass 1: logits (attention) + new_e_feat via WMMA; segment-max atomics.
// One wave = 16 edges x 64 outputs. K-loop over concat [Nd(128)|Ef(64)|Ns(128)].
// ---------------------------------------------------------------------------
__global__ __launch_bounds__(256) void k_edge_pass1(
    const __bf16* __restrict__ nb, const __bf16* __restrict__ eb,
    const int* __restrict__ src, const int* __restrict__ dst,
    const __bf16* __restrict__ WaP, const __bf16* __restrict__ WeP,
    const __bf16* __restrict__ WeeP, const float* __restrict__ prelu_a,
    unsigned int* __restrict__ maxenc, float* __restrict__ out_e) {
  __shared__ uint4 sWa4[2560];   // 40 KB: packed W_a (B fragments)
  __shared__ int sDst[8][16];

  const int tid = threadIdx.x;
  {
    const uint4* g = (const uint4*)WaP;
    for (int i = tid; i < 2560; i += 256) sWa4[i] = g[i];
  }
  const int wave = tid >> 5, lane = tid & 31;
  const int lm = lane & 15, hi = lane >> 4;
  const long ebase = (long)(blockIdx.x * 8 + wave) * 16;

  const int sIdx = src[ebase + lm];
  const int dIdx = dst[ebase + lm];
  if (hi == 0) sDst[wave][lm] = dIdx;
  __syncthreads();

  const float pa = prelu_a[0];

  v8f acc_l[4] = {};   // logits tile (16x64)
  v8f acc_e[4] = {};   // new_e_feat tile (16x64)

#pragma unroll
  for (int kt = 0; kt < 10; ++kt) {
    const __bf16* arow;
    const __bf16* w2; int kt2;
    if (kt < 4)      { arow = nb + (long)dIdx * 128 + kt * 32;        w2 = WeP;  kt2 = kt;     }
    else if (kt < 6) { arow = eb + (ebase + lm) * 64 + (kt - 4) * 32; w2 = WeeP; kt2 = kt - 4; }
    else             { arow = nb + (long)sIdx * 128 + (kt - 6) * 32;  w2 = WeP;  kt2 = kt - 6; }

    // A fragment: lane holds K in [16hi,16hi+16) (contiguous, matches packed B)
    ABu a;
    a.q[0] = *(const uint4*)(arow + 16 * hi);
    a.q[1] = *(const uint4*)(arow + 16 * hi + 8);

#pragma unroll
    for (int nt = 0; nt < 4; ++nt) {
      ABu bA, bE;
      const uint4* pA = sWa4 + ((kt * 2 + hi) * 64 + nt * 16 + lm) * 2;
      bA.q[0] = pA[0]; bA.q[1] = pA[1];
      const uint4* pE = (const uint4*)w2 + ((kt2 * 2 + hi) * 64 + nt * 16 + lm) * 2;
      bE.q[0] = pE[0]; bE.q[1] = pE[1];
      acc_l[nt] = __builtin_amdgcn_wmma_f32_16x16x32_bf16(
          false, a.v, false, bA.v, (short)0, acc_l[nt], false, false);
      acc_e[nt] = __builtin_amdgcn_wmma_f32_16x16x32_bf16(
          false, a.v, false, bE.v, (short)0, acc_e[nt], false, false);
    }
  }

#pragma unroll
  for (int nt = 0; nt < 4; ++nt) {
#pragma unroll
    for (int i = 0; i < 8; ++i) {
      const int r = i + 8 * hi;          // C layout: vgpr i -> row i (+8 for hi half)
      const int col = nt * 16 + lm;
      out_e[(ebase + r) * 64 + col] = acc_e[nt][i];
      float lv = acc_l[nt][i];
      lv = (lv >= 0.0f) ? lv : pa * lv;  // PReLU
      atomicMax(&maxenc[(long)sDst[wave][r] * 64 + col], enc_f32(lv));
    }
  }
}

// ---------------------------------------------------------------------------
// Edge pass 2: recompute logits + unatt, exp(l - max), atomic denom/num sums.
// ---------------------------------------------------------------------------
__global__ __launch_bounds__(256) void k_edge_pass2(
    const __bf16* __restrict__ nb, const __bf16* __restrict__ eb,
    const int* __restrict__ src, const int* __restrict__ dst,
    const __bf16* __restrict__ WaP, const __bf16* __restrict__ WtP,
    const float* __restrict__ prelu_a, const unsigned int* __restrict__ maxenc,
    float* __restrict__ denom, float* __restrict__ num) {
  __shared__ uint4 sWa4[2560];
  __shared__ int sDst[8][16];

  const int tid = threadIdx.x;
  {
    const uint4* g = (const uint4*)WaP;
    for (int i = tid; i < 2560; i += 256) sWa4[i] = g[i];
  }
  const int wave = tid >> 5, lane = tid & 31;
  const int lm = lane & 15, hi = lane >> 4;
  const long ebase = (long)(blockIdx.x * 8 + wave) * 16;

  const int sIdx = src[ebase + lm];
  const int dIdx = dst[ebase + lm];
  if (hi == 0) sDst[wave][lm] = dIdx;
  __syncthreads();

  const float pa = prelu_a[0];

  v8f acc_l[4] = {};   // logits
  v8f acc_u[4] = {};   // unattended messages

#pragma unroll
  for (int kt = 0; kt < 10; ++kt) {
    const __bf16* arow;
    if (kt < 4)      arow = nb + (long)dIdx * 128 + kt * 32;
    else if (kt < 6) arow = eb + (ebase + lm) * 64 + (kt - 4) * 32;
    else             arow = nb + (long)sIdx * 128 + (kt - 6) * 32;

    ABu a;
    a.q[0] = *(const uint4*)(arow + 16 * hi);
    a.q[1] = *(const uint4*)(arow + 16 * hi + 8);

#pragma unroll
    for (int nt = 0; nt < 4; ++nt) {
      ABu bA, bT;
      const uint4* pA = sWa4 + ((kt * 2 + hi) * 64 + nt * 16 + lm) * 2;
      bA.q[0] = pA[0]; bA.q[1] = pA[1];
      const uint4* pT = (const uint4*)WtP + ((kt * 2 + hi) * 64 + nt * 16 + lm) * 2;
      bT.q[0] = pT[0]; bT.q[1] = pT[1];
      acc_l[nt] = __builtin_amdgcn_wmma_f32_16x16x32_bf16(
          false, a.v, false, bA.v, (short)0, acc_l[nt], false, false);
      acc_u[nt] = __builtin_amdgcn_wmma_f32_16x16x32_bf16(
          false, a.v, false, bT.v, (short)0, acc_u[nt], false, false);
    }
  }

#pragma unroll
  for (int nt = 0; nt < 4; ++nt) {
#pragma unroll
    for (int i = 0; i < 8; ++i) {
      const int r = i + 8 * hi;
      const int col = nt * 16 + lm;
      const long o = (long)sDst[wave][r] * 64 + col;
      float lv = acc_l[nt][i];
      lv = (lv >= 0.0f) ? lv : pa * lv;
      const float ex = __expf(lv - dec_f32(maxenc[o]));
      atomicAdd(&denom[o], ex);
      atomicAdd(&num[o], ex * acc_u[nt][i]);
    }
  }
}

// ---------------------------------------------------------------------------
// Node epilogue: new_n_feat = has_msg ? num/denom + b_T : 0
// (denom > 0 iff node has edges: at least one ex == 1 per segment)
// ---------------------------------------------------------------------------
__global__ __launch_bounds__(256) void k_node(const float* __restrict__ dn,
                                              const float* __restrict__ nm,
                                              const float* __restrict__ bT,
                                              float* __restrict__ outn) {
  const int i = blockIdx.x * blockDim.x + threadIdx.x;
  if (i < NODE_ELEMS) {
    const float d = dn[i];
    outn[i] = (d > 0.0f) ? (nm[i] / d + bT[i & 63]) : 0.0f;
  }
}

// ---------------------------------------------------------------------------
// launch
// ---------------------------------------------------------------------------
extern "C" void kernel_launch(void* const* d_in, const int* in_sizes, int n_in,
                              void* d_out, int out_size, void* d_ws, size_t ws_size,
                              hipStream_t stream) {
  const float* n_feat  = (const float*)d_in[0];
  const float* e_feat  = (const float*)d_in[1];
  const int*   src     = (const int*)d_in[2];
  const int*   dst     = (const int*)d_in[3];
  const float* W_a     = (const float*)d_in[4];
  const float* W_T     = (const float*)d_in[5];
  const float* b_T     = (const float*)d_in[6];
  const float* W_e     = (const float*)d_in[7];
  const float* W_ee    = (const float*)d_in[8];
  const float* prelu_a = (const float*)d_in[9];

  // workspace layout (all 256B-aligned offsets), total ~153.7 MB
  char* ws = (char*)d_ws;
  unsigned* maxenc = (unsigned*)(ws + 0);            // 3.2M u32
  float*    denom  = (float*)(ws + 12800000);        // 3.2M f32
  float*    num    = (float*)(ws + 25600000);        // 3.2M f32
  __bf16*   nb     = (__bf16*)(ws + 38400000);       // 50000x128 bf16
  __bf16*   eb     = (__bf16*)(ws + 51200000);       // 800000x64 bf16
  __bf16*   WaP    = (__bf16*)(ws + 153600000);      // 20480 bf16
  __bf16*   WtP    = (__bf16*)(ws + 153640960);      // 20480 bf16
  __bf16*   WeP    = (__bf16*)(ws + 153681920);      // 8192 bf16
  __bf16*   WeeP   = (__bf16*)(ws + 153698304);      // 4096 bf16

  float* out_n = (float*)d_out;                      // 50000x64
  float* out_e = out_n + (long)NODE_ELEMS;           // 800000x64

  k_init<<<(NODE_ELEMS + 255) / 256, 256, 0, stream>>>(maxenc, denom, num);
  k_cvt<<<(6400000 / 4 + 255) / 256, 256, 0, stream>>>(n_feat, nb, 6400000);
  k_cvt<<<(51200000 / 4 + 255) / 256, 256, 0, stream>>>(e_feat, eb, 51200000);
  k_pack<<<80, 256, 0, stream>>>(W_a, W_T, W_e, W_ee, WaP, WtP, WeP, WeeP);

  // 800000 edges = 6250 blocks * 8 waves * 16 edges (exact, no tail)
  k_edge_pass1<<<6250, 256, 0, stream>>>(nb, eb, src, dst, WaP, WeP, WeeP,
                                         prelu_a, maxenc, out_e);
  k_edge_pass2<<<6250, 256, 0, stream>>>(nb, eb, src, dst, WaP, WtP,
                                         prelu_a, maxenc, denom, num);
  k_node<<<(NODE_ELEMS + 255) / 256, 256, 0, stream>>>(denom, num, b_T, out_n);
}
